// GraphAttentionLayer_37941741092941
// MI455X (gfx1250) — matrix operand order repaired and checked by reading, not compiled, not run on hardware
//
#include <hip/hip_runtime.h>

typedef unsigned short u16;
typedef unsigned int   u32;
typedef __attribute__((ext_vector_type(16))) __bf16 v16bf;
typedef __attribute__((ext_vector_type(8)))  float  v8f;

#define B_TOT   2048
#define M_NB    64
#define F_IN    512
#define H_HEADS 8
#define F_OUT   128
#define ROWS    8      // b-rows per workgroup
#define NBS     520    // nbr LDS row stride (ushorts, padded)
#define ABSS    520    // A-buf LDS row stride (ushorts, padded)
#define SXS     516    // self_x LDS row stride (f32, padded)

__device__ __forceinline__ u16 f2bf(float x) {
  u32 u = __float_as_uint(x);
  u += 0x7FFFu + ((u >> 16) & 1u);      // round-to-nearest-even
  return (u16)(u >> 16);
}

// ---- prep 1: w_self[h,f] = sum_o W[h,f,o]*a_self[h,o]; same for a_nb ----
__global__ void gat_prep_wvec(const float* __restrict__ W,
                              const float* __restrict__ a_self,
                              const float* __restrict__ a_nb,
                              float* __restrict__ wself,
                              float* __restrict__ wnb) {
  int id = blockIdx.x * 256 + threadIdx.x;        // over H*F_IN = 4096
  if (id >= H_HEADS * F_IN) return;
  int h = id >> 9;
  const float* wr = W + (size_t)id * F_OUT;       // W[h][f][:]
  const float* as = a_self + h * F_OUT;
  const float* an = a_nb + h * F_OUT;
  float s1 = 0.f, s2 = 0.f;
#pragma unroll 8
  for (int o = 0; o < F_OUT; ++o) {
    float w = wr[o];
    s1 += w * as[o];
    s2 += w * an[o];
  }
  wself[id] = s1;
  wnb[id]   = s2;
}

// ---- prep 2: swizzle W (f32) -> bf16 B-matrix fragments in contiguous-k order ----
// Wfrag[((h*16+kk)*8+nt)*32 + lane][e]  where  k = kk*32 + (lane>=16?16:0) + e,
// o = nt*16 + lane%16.  Total 8*16*8*32*16 = 524288 bf16 = 1 MB.
__global__ void gat_prep_wfrag(const float* __restrict__ W, u16* __restrict__ wfrag) {
  int idx = blockIdx.x * 256 + threadIdx.x;       // 524288 total
  int e  = idx & 15;
  int l  = (idx >> 4) & 31;
  int nt = (idx >> 9) & 7;
  int kk = (idx >> 12) & 15;
  int h  = idx >> 16;
  int k  = kk * 32 + ((l >= 16) ? 16 : 0) + e;
  int o  = nt * 16 + (l & 15);
  wfrag[idx] = f2bf(W[((size_t)h * F_IN + k) * F_OUT + o]);
}

// ---- main: one WG per 8 batch rows; 8 waves = 8 heads ----
__global__ void __launch_bounds__(256)
gat_main(const float* __restrict__ self_x, const float* __restrict__ nbx,
         const float* __restrict__ wself,  const float* __restrict__ wnb,
         const u16* __restrict__ wfrag,    float* __restrict__ out) {
  extern __shared__ char smem[];
  float* s_wslf = (float*)(smem);              // [8][512]      16384 B
  float* s_wnb  = (float*)(smem + 16384);      // [8][512]      16384 B
  float* s_sx   = (float*)(smem + 32768);      // [8][516]      16512 B
  float* s_esf  = (float*)(smem + 49280);      // [8][8]          256 B
  float* s_enb  = (float*)(smem + 49536);      // [8][64]        2048 B
  u16*   s_alp  = (u16*)(smem + 51584);        // [16][64] bf16  2048 B (rows 8..15 = 0)
  u16*   s_wnbf = (u16*)(smem + 53632);        // [16kk][32][16] 16384 B (prestaged B frags)
  u16*   s_nbr  = (u16*)(smem + 70016);        // [64][520]     66560 B
  u16*   s_ab   = (u16*)(smem + 136576);       // [8][16][520] 133120 B (rows 8..15 = 0)
                                               // total 269696 B < 320 KB

  const int t    = threadIdx.x;
  const int lane = t & 31;
  const int wid  = t >> 5;
  const int b0   = blockIdx.x * ROWS;

  // phase 0a: stage w vectors + self rows
  for (int i = t; i < H_HEADS * F_IN; i += 256) {
    s_wslf[i] = wself[i];
    s_wnb[i]  = wnb[i];
  }
  for (int i = t; i < ROWS * F_IN; i += 256) {
    int r = i >> 9, f = i & 511;
    s_sx[r * SXS + f] = self_x[(size_t)(b0 + r) * F_IN + f];
  }
  __syncthreads();
  // phase 0b: e_self dots
  if (t < ROWS * H_HEADS) {
    int r = t >> 3, h = t & 7;
    const float* xr = s_sx + r * SXS;
    const float* wr = s_wslf + h * F_IN;
    float s = 0.f;
#pragma unroll 8
    for (int f = 0; f < F_IN; ++f) s += xr[f] * wr[f];
    s_esf[r * 8 + h] = s;
  }
  // phase 0c: prestage w_nb B fragments in fragment-linear bf16 order
  for (int idx = t; idx < 16 * 32 * 16; idx += 256) {
    int e  = idx & 15;
    int l  = (idx >> 4) & 31;
    int kk = idx >> 9;
    int n  = l & 15;
    int k  = kk * 32 + ((l >= 16) ? 16 : 0) + e;
    s_wnbf[idx] = (n < H_HEADS) ? f2bf(s_wnb[n * F_IN + k]) : (u16)0;
  }
  // phase 0d: zero pad rows (8..15) of alpha and A-buf so gathers are branch-free
  for (int i = t; i < 8 * 64; i += 256) s_alp[8 * 64 + i] = 0;
  for (int i = t; i < H_HEADS * 8 * F_IN; i += 256) {
    int h = i >> 12;            // /4096
    int r = (i >> 9) & 7;       // padded row 8..15
    int f = i & 511;
    s_ab[(h * 16 + 8 + r) * ABSS + f] = 0;
  }
  __syncthreads();

  for (int bi = 0; bi < ROWS; ++bi) {
    const int b = b0 + bi;
    // --- stream neighbor_x[b] (64x512 f32) -> LDS bf16, read-once from HBM ---
    {
      const float4* src = (const float4*)(nbx + (size_t)b * M_NB * F_IN);
#pragma unroll 4
      for (int i = 0; i < 32; ++i) {
        int j = i * 256 + t;                    // 8192 float4s
        float4 v = src[j];
        int m  = j >> 7;
        int f4 = (j & 127) << 2;
        ushort4 pk;
        pk.x = f2bf(v.x); pk.y = f2bf(v.y); pk.z = f2bf(v.z); pk.w = f2bf(v.w);
        *(ushort4*)(s_nbr + m * NBS + f4) = pk;
      }
    }
    __syncthreads();

    // --- e_nb = nbr(64x512) @ w_nb^T(512x16; 8 cols used), WMMA, waves 0..3 ---
    if (wid < 4) {
      const int mt = wid;
      v8f acc = {};
      for (int kk = 0; kk < 16; ++kk) {
        union { u16 u[16]; v16bf v; } a;
        {
          int m = mt * 16 + (lane & 15);
          const u16* ar = s_nbr + m * NBS + kk * 32 + ((lane >= 16) ? 16 : 0);
#pragma unroll
          for (int e = 0; e < 16; ++e) a.u[e] = ar[e];
        }
        v16bf bv = *(const v16bf*)(s_wnbf + (kk * 32 + lane) * 16);
        acc = __builtin_amdgcn_wmma_f32_16x16x32_bf16(false, a.v, false, bv,
                                                      (short)0, acc, false, false);
      }
      int n = lane & 15;
      if (n < H_HEADS) {
        int mb = mt * 16 + ((lane < 16) ? 0 : 8);
#pragma unroll
        for (int v = 0; v < 8; ++v) s_enb[n * 64 + mb + v] = acc[v];
      }
    }
    __syncthreads();

    // --- softmax over 64 neighbors; wave `wid` owns head `wid` (wave32 shuffles) ---
    {
      const int h = wid;
      float es = s_esf[bi * 8 + h];
      float v0 = es + s_enb[h * 64 + lane];
      float v1 = es + s_enb[h * 64 + lane + 32];
      v0 = (v0 > 0.f) ? v0 : 0.2f * v0;
      v1 = (v1 > 0.f) ? v1 : 0.2f * v1;
      float mx = fmaxf(v0, v1);
#pragma unroll
      for (int off = 16; off >= 1; off >>= 1) mx = fmaxf(mx, __shfl_xor(mx, off, 32));
      float e0 = __expf(v0 - mx), e1 = __expf(v1 - mx);
      float s = e0 + e1;
#pragma unroll
      for (int off = 16; off >= 1; off >>= 1) s += __shfl_xor(s, off, 32);
      float inv = 1.f / s;
      s_alp[h * 64 + lane]      = f2bf(e0 * inv);   // bf16 alpha, A-gather-friendly
      s_alp[h * 64 + lane + 32] = f2bf(e1 * inv);
    }
    __syncthreads();

    // --- xbar(16h x 512f) = alpha(16x64) @ nbr(64x512); wave w owns nt=4w..4w+3 ---
    {
      v8f acc[4] = {};
      for (int kk = 0; kk < 2; ++kk) {
        union { u16 u[16]; v16bf v; } a;
        {
          const u16* ar = s_alp + (lane & 15) * 64 + kk * 32 + ((lane >= 16) ? 16 : 0);
#pragma unroll
          for (int e = 0; e < 16; ++e) a.u[e] = ar[e];
        }
#pragma unroll
        for (int j = 0; j < 4; ++j) {
          int nt = wid * 4 + j;
          union { u16 u[16]; v16bf v; } bm;
          int fcol = nt * 16 + (lane & 15);
          int kb   = kk * 32 + ((lane >= 16) ? 16 : 0);
#pragma unroll
          for (int e = 0; e < 16; ++e) bm.u[e] = s_nbr[(kb + e) * NBS + fcol];
          acc[j] = __builtin_amdgcn_wmma_f32_16x16x32_bf16(false, a.v, false, bm.v,
                                                           (short)0, acc[j], false, false);
        }
      }
      if (lane < 16) {               // rows h = v (0..7) valid
#pragma unroll
        for (int j = 0; j < 4; ++j) {
          int f = (wid * 4 + j) * 16 + lane;
#pragma unroll
          for (int v = 0; v < 8; ++v) {
            float val = acc[j][v] + s_sx[bi * SXS + f];   // A' = self + xbar
            s_ab[(v * 16 + bi) * ABSS + f] = f2bf(val);
          }
        }
      }
    }
    __syncthreads();
  }

  // --- phase B: out[b, h*128+:] = relu( A'(16x512) @ W[h](512x128) ), wave=head ---
  {
    const int h = wid;
    v8f acc[8] = {};
    for (int kk = 0; kk < 16; ++kk) {
      union { u16 u[16]; v16bf v; } a;
      {
        const u16* ar = s_ab + (h * 16 + (lane & 15)) * ABSS + kk * 32 +
                        ((lane >= 16) ? 16 : 0);
#pragma unroll
        for (int e = 0; e < 16; ++e) a.u[e] = ar[e];
      }
      const u16* wk = wfrag + (size_t)((h * 16 + kk) * 8) * 512;
#pragma unroll
      for (int nt = 0; nt < 8; ++nt) {
        v16bf bv = *((const v16bf*)(wk + nt * 512) + lane);  // 32B/lane, pre-swizzled
        acc[nt] = __builtin_amdgcn_wmma_f32_16x16x32_bf16(false, a.v, false, bv,
                                                          (short)0, acc[nt], false, false);
      }
    }
    if (lane < 16) {                 // rows m = v (0..7) valid
#pragma unroll
      for (int nt = 0; nt < 8; ++nt) {
        int col = h * F_OUT + nt * 16 + lane;
#pragma unroll
        for (int v = 0; v < 8; ++v)
          out[(size_t)(b0 + v) * (H_HEADS * F_OUT) + col] = fmaxf(acc[nt][v], 0.f);
      }
    }
  }
}

extern "C" void kernel_launch(void* const* d_in, const int* in_sizes, int n_in,
                              void* d_out, int out_size, void* d_ws, size_t ws_size,
                              hipStream_t stream) {
  (void)in_sizes; (void)n_in; (void)out_size; (void)ws_size;
  const float* self_x = (const float*)d_in[0];
  const float* nbx    = (const float*)d_in[1];
  const float* W      = (const float*)d_in[2];
  const float* a_self = (const float*)d_in[3];
  const float* a_nb   = (const float*)d_in[4];
  float* out = (float*)d_out;

  float* wself = (float*)d_ws;                       // 8*512 f32
  float* wnb   = wself + H_HEADS * F_IN;             // 8*512 f32
  u16*   wfrag = (u16*)(wnb + H_HEADS * F_IN);       // 524288 bf16 (1 MB)

  gat_prep_wvec<<<(H_HEADS * F_IN + 255) / 256, 256, 0, stream>>>(W, a_self, a_nb, wself, wnb);
  gat_prep_wfrag<<<(H_HEADS * 16 * 8 * 32 * 16) / 256, 256, 0, stream>>>(W, wfrag);

  constexpr size_t SMEM = 269696;  // < 320 KB/WGP
  hipFuncSetAttribute(reinterpret_cast<const void*>(gat_main),
                      hipFuncAttributeMaxDynamicSharedMemorySize, (int)SMEM);
  gat_main<<<B_TOT / ROWS, 256, SMEM, stream>>>(self_x, nbx, wself, wnb, wfrag, out);
}
